// NeuralODECNN_54614804135958
// MI455X (gfx1250) — matrix operand order repaired and checked by reading, not compiled, not run on hardware
//
#include <hip/hip_runtime.h>
#include <hip/hip_bf16.h>
#include <math.h>

typedef __attribute__((ext_vector_type(16))) __bf16 bf16x16;
typedef __attribute__((ext_vector_type(8)))  float  f32x8;

#define B_    128
#define HC_   32
#define HHC_  128
#define P_    4
#define HW_   1024           // 32*32
#define NPIX  (B_ * HW_)     // 131072 pixels
#define NZ    (NPIX * HC_)   // 4194304 state elements

#define LOG2E_F 1.4426950408889634f
#define LN2_F   0.6931471805599453f

// A-matrix (16-bit, 16x32) element->K mapping (contiguous, matches B):
// lane L (0-15): M=L, elements 0..15 hold K=0..15
// lane L+16   : M=L, elements 0..15 hold K=16..31
__device__ __forceinline__ int kmapA(int i, int lane) {
  return i + ((lane >= 16) ? 16 : 0);
}
// B-matrix (16-bit, 32x16): lanes 0-15 hold K=0..15 (N=lane), lanes 16-31 K=16..31
__device__ __forceinline__ int koff16(int lane) { return (lane >= 16) ? 16 : 0; }

// Raw-HW softplus: exp2 arg <= 0 (no overflow), log2 arg in [1,2] (never
// denormal) -> safe to use unguarded v_exp_f32 / v_log_f32.
__device__ __forceinline__ float softplus_fast(float x) {
  float e = __builtin_amdgcn_exp2f(-LOG2E_F * fabsf(x));
  return fmaxf(x, 0.f) + LN2_F * __builtin_amdgcn_logf(1.f + e);
}
// Raw-HW tanh: 1+e in [1,2] -> unguarded v_rcp_f32 is exact enough.
__device__ __forceinline__ float tanh_fast(float x) {
  float e = __builtin_amdgcn_exp2f(-2.f * LOG2E_F * fabsf(x));
  float t = (1.f - e) * __builtin_amdgcn_rcpf(1.f + e);
  return copysignf(t, x);
}

// ---------------------------------------------------------------------------
// Pack weights into WMMA A-fragment order (one contiguous 32B chunk per lane).
// ---------------------------------------------------------------------------
__global__ __launch_bounds__(256) void pack_kernel(
    const float* __restrict__ W1, const float* __restrict__ W2,
    const float* __restrict__ W3,
    __bf16* __restrict__ A1, float* __restrict__ w1t,
    __bf16* __restrict__ A2, __bf16* __restrict__ A3) {
  int idx = blockIdx.x * 256 + threadIdx.x;
  if (idx < 512) {  // t-channel coefficient of W1 (folded into bias at runtime)
    int p = idx >> 7, oc = idx & 127;
    w1t[idx] = W1[(p * 128 + oc) * 33];
  }
  if (idx < 36864) {  // A2: [p][cg][tap][octile][lane][16]
    int lane = idx & 31;
    int r = idx >> 5;
    int ot = r & 7;  r >>= 3;
    int tap = r % 9; r /= 9;
    int cg = r & 3;  int p = r >> 2;
    int oc = ot * 16 + (lane & 15);
    int ky = tap / 3, kx = tap % 3;
    for (int i = 0; i < 16; ++i) {
      int cin = cg * 32 + kmapA(i, lane);
      A2[(size_t)idx * 16 + i] =
          (__bf16)W2[((((p * 128 + oc) * 128) + cin) * 3 + ky) * 3 + kx];
    }
  } else if (idx < 37888) {  // A1: [p][octile][lane][16], K = z-channels (skip t col)
    int j = idx - 36864;
    int lane = j & 31; int ot = (j >> 5) & 7; int p = j >> 8;
    int oc = ot * 16 + (lane & 15);
    for (int i = 0; i < 16; ++i)
      A1[j * 16 + i] = (__bf16)W1[(p * 128 + oc) * 33 + 1 + kmapA(i, lane)];
  } else if (idx < 38912) {  // A3: [p][kb][octile][lane][16]
    int j = idx - 37888;
    int lane = j & 31; int ot = (j >> 5) & 1; int kb = (j >> 6) & 3; int p = j >> 8;
    int oc = ot * 16 + (lane & 15);
    for (int i = 0; i < 16; ++i)
      A3[j * 16 + i] = (__bf16)W3[(p * 32 + oc) * 128 + kb * 32 + kmapA(i, lane)];
  }
}

// ---------------------------------------------------------------------------
// Augment: z[pix][oc] = aug_W @ x  (3 -> 32 channels, pixel-major layout)
// ---------------------------------------------------------------------------
__global__ __launch_bounds__(256) void augment_kernel(
    const float* __restrict__ x, const float* __restrict__ augW,
    const float* __restrict__ augb, float* __restrict__ z) {
  int i = blockIdx.x * 256 + threadIdx.x;  // NZ threads
  int oc = i & 31; int pix = i >> 5;
  int b = pix >> 10, s = pix & 1023;
  float v = augb[oc];
  for (int c = 0; c < 3; ++c)
    v += x[(b * 3 + c) * 1024 + s] * augW[oc * 3 + c];
  z[i] = v;
}

// ---------------------------------------------------------------------------
// RK4 elementwise state update + bf16 stage-input generation (fused, x4 vec).
// mode 0: acc=z, zin=bf16(z)        (start of step)
// mode 1: acc+=wacc*k, zin=bf16(z+cst*k)
// mode 2: z=acc+wacc*k, acc=z, zin=bf16(z)   (end of step)
// ---------------------------------------------------------------------------
__global__ __launch_bounds__(256) void rk_stage_kernel(
    float* __restrict__ z, float* __restrict__ acc, const float* __restrict__ k,
    __bf16* __restrict__ zin, float wacc, float cst, int mode) {
  int i = (blockIdx.x * 256 + threadIdx.x) * 4;
  __bf16 o[4] __attribute__((aligned(8)));
  if (mode == 0) {
    float4 zv = *(const float4*)(z + i);
    *(float4*)(acc + i) = zv;
    o[0] = (__bf16)zv.x; o[1] = (__bf16)zv.y;
    o[2] = (__bf16)zv.z; o[3] = (__bf16)zv.w;
  } else if (mode == 1) {
    float4 kv = *(const float4*)(k + i);
    float4 av = *(const float4*)(acc + i);
    float4 zv = *(const float4*)(z + i);
    av.x += wacc * kv.x; av.y += wacc * kv.y;
    av.z += wacc * kv.z; av.w += wacc * kv.w;
    *(float4*)(acc + i) = av;
    o[0] = (__bf16)(zv.x + cst * kv.x); o[1] = (__bf16)(zv.y + cst * kv.y);
    o[2] = (__bf16)(zv.z + cst * kv.z); o[3] = (__bf16)(zv.w + cst * kv.w);
  } else {
    float4 kv = *(const float4*)(k + i);
    float4 av = *(const float4*)(acc + i);
    av.x += wacc * kv.x; av.y += wacc * kv.y;
    av.z += wacc * kv.z; av.w += wacc * kv.w;
    *(float4*)(z + i) = av;
    *(float4*)(acc + i) = av;
    o[0] = (__bf16)av.x; o[1] = (__bf16)av.y;
    o[2] = (__bf16)av.z; o[3] = (__bf16)av.w;
  }
  *(int2*)(zin + i) = *(int2*)o;
}

// ---------------------------------------------------------------------------
// GEMM1: h1 = softplus(W1[:,1:] @ zin + (b1 + W1[:,0]*t))   33->128, K=32
// Block = 8 waves; wave w = out-channel tile w; 4 pixel tiles per block,
// A fragment and bias kept register-resident across tiles.
// ---------------------------------------------------------------------------
__global__ __launch_bounds__(256) void gemm1_kernel(
    const __bf16* __restrict__ zin, const __bf16* __restrict__ A1p,
    const float* __restrict__ b1p, const float* __restrict__ w1tp,
    float t, __bf16* __restrict__ h1) {
  __shared__ __bf16 outb[64 * 128];  // 16KB transpose buffer
  int tid = threadIdx.x, wave = tid >> 5, lane = tid & 31;
  long pixbase0 = (long)blockIdx.x * 64;

  bf16x16 a = *(const bf16x16*)(A1p + (wave * 32 + lane) * 16);
  int ocb = wave * 16 + ((lane >= 16) ? 8 : 0);
  float bias[8];
  for (int r = 0; r < 8; ++r) bias[r] = b1p[ocb + r] + w1tp[ocb + r] * t;

  for (int pt = 0; pt < 4; ++pt) {
    long pixbase = pixbase0 + pt * 16;
    bf16x16 bf =
        *(const bf16x16*)(zin + (pixbase + (lane & 15)) * 32 + koff16(lane));
    f32x8 c = {};
    c = __builtin_amdgcn_wmma_f32_16x16x32_bf16(false, a, false, bf, (short)0, c,
                                                false, false);
    __bf16 res[8] __attribute__((aligned(16)));
    for (int r = 0; r < 8; ++r)
      res[r] = (__bf16)softplus_fast(c[r] + bias[r]);
    *(int4*)&outb[(pt * 16 + (lane & 15)) * 128 + ocb] = *(int4*)res;
  }
  __syncthreads();
  int4* gout = (int4*)(h1 + pixbase0 * 128);  // 8192 halves = 1024 int4
  for (int j = 0; j < 4; ++j)
    gout[j * 256 + tid] = ((const int4*)outb)[j * 256 + tid];  // coalesced
}

// ---------------------------------------------------------------------------
// CONV2 (the hot loop): 3x3 SAME conv 128->128 as implicit GEMM, K=1152.
// Block = (image b, 4 output rows) -> 128 pixels x 128 oc.
// LDS halo tile [6 rows][34 x][32 c] staged once per 32-channel group,
// reused by 9 taps x 8 waves. 2304 WMMAs per block.
// ---------------------------------------------------------------------------
__global__ __launch_bounds__(256) void conv2_kernel(
    const __bf16* __restrict__ h1, const __bf16* __restrict__ A2p,
    const float* __restrict__ b2p, __bf16* __restrict__ h2) {
  __shared__ __bf16 tile[204 * 32];   // 6*34 halo positions x 32 channels
  __shared__ __bf16 outb[128 * 128];  // output transpose buffer
  int tid = threadIdx.x, wave = tid >> 5, lane = tid & 31;
  int b = blockIdx.x >> 3, ybase = (blockIdx.x & 7) * 4;

  f32x8 zerov = {};
  f32x8 acc[8];
  for (int t = 0; t < 8; ++t) acc[t] = zerov;

  for (int cg = 0; cg < 4; ++cg) {
    __syncthreads();  // previous tile reads complete
    for (int pr = tid; pr < 204; pr += 256) {
      int row = pr / 34, xc = pr % 34;
      int y = ybase - 1 + row, xx = xc - 1;
      int4* dst = (int4*)&tile[pr * 32];
      if ((unsigned)y < 32u && (unsigned)xx < 32u) {
        const int4* src =
            (const int4*)(h1 + ((long)(b * 1024 + y * 32 + xx)) * 128 + cg * 32);
        dst[0] = src[0]; dst[1] = src[1]; dst[2] = src[2]; dst[3] = src[3];
        if (cg < 3)  // pull next channel group toward the caches
          __builtin_prefetch((const char*)src + 64, 0, 1);
      } else {
        int4 zz = {0, 0, 0, 0};
        dst[0] = zz; dst[1] = zz; dst[2] = zz; dst[3] = zz;
      }
    }
    __syncthreads();

    int ko = koff16(lane), ln = lane & 15;
    for (int tap = 0; tap < 9; ++tap) {
      bf16x16 a = *(const bf16x16*)(A2p +
          (size_t)(((cg * 9 + tap) * 8 + wave) * 32 + lane) * 16);
      int dy = tap / 3, dx = tap % 3;
      for (int t = 0; t < 8; ++t) {
        int row = (t >> 1) + dy;                 // halo row (ybase-1 offset folded)
        int xc = (t & 1) * 16 + ln + dx;         // halo x
        bf16x16 bf = *(const bf16x16*)&tile[(row * 34 + xc) * 32 + ko];
        acc[t] = __builtin_amdgcn_wmma_f32_16x16x32_bf16(
            false, a, false, bf, (short)0, acc[t], false, false);
      }
    }
  }

  // bias + softplus, transpose through LDS, coalesced b128 stores
  int ocb = wave * 16 + ((lane >= 16) ? 8 : 0);
  for (int t = 0; t < 8; ++t) {
    __bf16 res[8] __attribute__((aligned(16)));
    for (int r = 0; r < 8; ++r)
      res[r] = (__bf16)softplus_fast(acc[t][r] + b2p[ocb + r]);
    int pix = (t >> 1) * 32 + (t & 1) * 16 + (lane & 15);
    *(int4*)&outb[pix * 128 + ocb] = *(int4*)res;
  }
  __syncthreads();
  long gbase = ((long)(b * 1024 + ybase * 32)) * 128;
  for (int j = 0; j < 8; ++j)
    ((int4*)(h2 + gbase))[tid * 8 + j] = ((const int4*)outb)[tid * 8 + j];
}

// ---------------------------------------------------------------------------
// GEMM3: k = tanh(W3 @ h2 + b3)   128->32, K=128 (4 WMMA), f32 output
// ---------------------------------------------------------------------------
__global__ __launch_bounds__(256) void gemm3_kernel(
    const __bf16* __restrict__ h2, const __bf16* __restrict__ A3p,
    const float* __restrict__ b3p, float* __restrict__ kout) {
  int tid = threadIdx.x, wave = tid >> 5, lane = tid & 31;
  int octile = wave & 1, ptile = wave >> 1;
  long pixbase = (long)blockIdx.x * 64 + ptile * 16;
  const __bf16* bbase = h2 + (pixbase + (lane & 15)) * 128 + koff16(lane);
  f32x8 c = {};
  for (int kb = 0; kb < 4; ++kb) {
    bf16x16 a = *(const bf16x16*)(A3p + ((kb * 2 + octile) * 32 + lane) * 16);
    bf16x16 bf = *(const bf16x16*)(bbase + kb * 32);
    c = __builtin_amdgcn_wmma_f32_16x16x32_bf16(false, a, false, bf, (short)0, c,
                                                false, false);
  }
  int ocb = octile * 16 + ((lane >= 16) ? 8 : 0);
  f32x8 res;
  for (int r = 0; r < 8; ++r) res[r] = tanh_fast(c[r] + b3p[ocb + r]);
  *(f32x8*)(kout + (pixbase + (lane & 15)) * 32 + ocb) = res;  // 32B aligned
}

// ---------------------------------------------------------------------------
// Readout + log-softmax loss + accuracy (one block per batch element)
// ---------------------------------------------------------------------------
__global__ __launch_bounds__(256) void readout_kernel(
    const float* __restrict__ z, const float* __restrict__ roW,
    const float* __restrict__ rob, const int* __restrict__ y,
    float* __restrict__ out) {
  __shared__ float red[10][256];
  int tid = threadIdx.x, b = blockIdx.x;
  float p[10];
  for (int o = 0; o < 10; ++o) p[o] = 0.f;
  for (int f = tid; f < 32768; f += 256) {  // reference feature order c*1024+s
    int c = f >> 10, s = f & 1023;
    float zv = z[((long)(b * 1024 + s)) * 32 + c];
    for (int o = 0; o < 10; ++o) p[o] += zv * roW[o * 32768 + f];
  }
  for (int o = 0; o < 10; ++o) red[o][tid] = p[o];
  __syncthreads();
  for (int st = 128; st > 0; st >>= 1) {
    if (tid < st)
      for (int o = 0; o < 10; ++o) red[o][tid] += red[o][tid + st];
    __syncthreads();
  }
  if (tid == 0) {
    float lg[10], mx = -1e30f;
    for (int o = 0; o < 10; ++o) { lg[o] = red[o][0] + rob[o]; mx = fmaxf(mx, lg[o]); }
    float se = 0.f;
    for (int o = 0; o < 10; ++o) se += __expf(lg[o] - mx);
    float lse = mx + __logf(se);
    int yb = y[b];
    atomicAdd(out, -(lg[yb] - lse) / 128.f);
    int am = 0; float bv = lg[0];
    for (int o = 1; o < 10; ++o) if (lg[o] > bv) { bv = lg[o]; am = o; }
    atomicAdd(out + 1, (am == yb) ? 1.f : 0.f);
  }
}

__global__ void init_out_kernel(float* out) {
  if (threadIdx.x < 2) out[threadIdx.x] = 0.f;
}

// ---------------------------------------------------------------------------
extern "C" void kernel_launch(void* const* d_in, const int* in_sizes, int n_in,
                              void* d_out, int out_size, void* d_ws, size_t ws_size,
                              hipStream_t stream) {
  const float* x    = (const float*)d_in[0];
  const int*   y    = (const int*)d_in[1];
  const float* augW = (const float*)d_in[2];
  const float* augb = (const float*)d_in[3];
  const float* W1   = (const float*)d_in[4];
  const float* b1   = (const float*)d_in[5];
  const float* W2   = (const float*)d_in[6];
  const float* b2   = (const float*)d_in[7];
  const float* W3   = (const float*)d_in[8];
  const float* b3   = (const float*)d_in[9];
  const float* roW  = (const float*)d_in[10];
  const float* rob  = (const float*)d_in[11];

  char* w = (char*)d_ws;
  auto alloc = [&](size_t bytes) {
    char* r = w; w += (bytes + 255) & ~(size_t)255; return r;
  };
  float*  z   = (float*)alloc((size_t)NZ * 4);
  float*  acc = (float*)alloc((size_t)NZ * 4);
  float*  kb  = (float*)alloc((size_t)NZ * 4);
  __bf16* zin = (__bf16*)alloc((size_t)NZ * 2);
  __bf16* h1  = (__bf16*)alloc((size_t)NPIX * 128 * 2);
  __bf16* h2  = (__bf16*)alloc((size_t)NPIX * 128 * 2);
  __bf16* A1  = (__bf16*)alloc((size_t)4 * 8 * 512 * 2);
  float*  w1t = (float*)alloc((size_t)4 * 128 * 4);
  __bf16* A2  = (__bf16*)alloc((size_t)4 * 36 * 8 * 512 * 2);
  __bf16* A3  = (__bf16*)alloc((size_t)4 * 4 * 2 * 512 * 2);

  pack_kernel<<<dim3(152), dim3(256), 0, stream>>>(W1, W2, W3, A1, w1t, A2, A3);
  augment_kernel<<<dim3(NZ / 256), dim3(256), 0, stream>>>(x, augW, augb, z);
  rk_stage_kernel<<<dim3(NZ / 1024), dim3(256), 0, stream>>>(z, acc, kb, zin,
                                                             0.f, 0.f, 0);

  const float w16 = 1.f / 6.f, w13 = 1.f / 3.f;
  for (int s = 0; s < 4; ++s) {
    const float ts[4]    = {(float)s, s + 0.5f, s + 0.5f, (float)(s + 1)};
    const int   ps[4]    = {s, s, s, (s + 1 > 3) ? 3 : s + 1};
    const float waccs[4] = {w16, w13, w13, w16};
    const float csts[4]  = {0.5f, 0.5f, 1.f, 0.f};
    const int   modes[4] = {1, 1, 1, 2};
    for (int st = 0; st < 4; ++st) {
      int p = ps[st]; float t = ts[st];
      gemm1_kernel<<<dim3(NPIX / 64), dim3(256), 0, stream>>>(
          zin, A1 + p * 8 * 512, b1 + p * 128, w1t + p * 128, t, h1);
      conv2_kernel<<<dim3(1024), dim3(256), 0, stream>>>(
          h1, A2 + (size_t)p * 36 * 8 * 512, b2 + p * 128, h2);
      gemm3_kernel<<<dim3(2048), dim3(256), 0, stream>>>(
          h2, A3 + p * 4 * 2 * 512, b3 + p * 32, kb);
      rk_stage_kernel<<<dim3(NZ / 1024), dim3(256), 0, stream>>>(
          z, acc, kb, zin, waccs[st], csts[st], modes[st]);
    }
  }

  init_out_kernel<<<dim3(1), dim3(32), 0, stream>>>((float*)d_out);
  readout_kernel<<<dim3(128), dim3(256), 0, stream>>>(z, roW, rob, y,
                                                      (float*)d_out);
}
